// Generator_59304908423691
// MI455X (gfx1250) — compile-verified, matrix operand order checked by reading, not run on hardware
//
#include <hip/hip_runtime.h>
#include <hip/hip_bf16.h>

// ---------------------------------------------------------------------------
// Types
// ---------------------------------------------------------------------------
typedef __attribute__((ext_vector_type(16))) __bf16 v16bf;
typedef __attribute__((ext_vector_type(8)))  float  v8f;
typedef __attribute__((ext_vector_type(4)))  float  v4f;
typedef __attribute__((ext_vector_type(4)))  unsigned int v4u;

#define T_FIN 16384          // final sample length per batch
#define NBATCH 4
#define CH 128
#define TBLK 64              // time samples per k_wn_gate block (4 N-tiles)

// scratch layout (bytes from d_ws base, all 4KB aligned)
#define OFF_X64    ((size_t)0)          // fp32 [4][64][128]
#define OFF_TMP64  ((size_t)131072)     // fp32 [4][64][128]
#define OFF_X256   ((size_t)262144)     // fp32 [4][256][128]
#define OFF_X1024  ((size_t)786432)     // fp32 [4][1024][128]
#define OFF_XBF    ((size_t)2883584)    // bf16 [4][16384][128]
#define OFF_GATED  ((size_t)19660800)   // bf16 [4][16384][128]
#define OFF_SKIPS  ((size_t)36438016)   // fp32 [4][16384][128]
#define OFF_WNP    ((size_t)69992448)   // bf16 packed WN conv weights (30*98304)
#define OFF_SKP    ((size_t)75890688)   // bf16 packed skip weights    (30*16384)
#define OFF_O1P    ((size_t)76873728)   // bf16 packed out1 weights    (16384)
#define OFF_O2P    ((size_t)76906496)   // bf16 packed out2 weights    (16384)

// ---------------------------------------------------------------------------
// bf16 helpers (RNE)
// ---------------------------------------------------------------------------
__device__ __forceinline__ unsigned short f2bf(float f) {
  unsigned int u = __float_as_uint(f);
  u = u + 0x7FFFu + ((u >> 16) & 1u);
  return (unsigned short)(u >> 16);
}
__device__ __forceinline__ float bf2f(unsigned short h) {
  return __uint_as_float(((unsigned int)h) << 16);
}
__device__ __forceinline__ v4u pack8(const float* f) {
  v4u r;
  r.x = (unsigned)f2bf(f[0]) | ((unsigned)f2bf(f[1]) << 16);
  r.y = (unsigned)f2bf(f[2]) | ((unsigned)f2bf(f[3]) << 16);
  r.z = (unsigned)f2bf(f[4]) | ((unsigned)f2bf(f[5]) << 16);
  r.w = (unsigned)f2bf(f[6]) | ((unsigned)f2bf(f[7]) << 16);
  return r;
}
__device__ __forceinline__ void unpack8(v4u v, float* f) {
  unsigned w[4] = {v.x, v.y, v.z, v.w};
  #pragma unroll
  for (int r = 0; r < 8; ++r) {
    unsigned word = w[r >> 1];
    unsigned short h = (r & 1) ? (unsigned short)(word >> 16)
                               : (unsigned short)(word & 0xFFFF);
    f[r] = bf2f(h);
  }
}

union Frag { v16bf v; v4u q[2]; };

__device__ __forceinline__ v8f wmma_bf16(v16bf a, v16bf b, v8f c) {
  // D = A(16x32 bf16) x B(32x16 bf16) + C(16x16 f32)
  return __builtin_amdgcn_wmma_f32_16x16x32_bf16(
      false, a, false, b, (short)0, c, false, false);
}

// B operand from an LDS tile stored [t_local][c(128)] bf16 (rowBase in elems).
// lane n (lane&15) holds column N=n; lane group hi adds +8 to K pattern.
// K per lane-half h: (h&7) + (h>=8 ? 16:0) + hi*8  -> two contiguous 8-half runs.
__device__ __forceinline__ v16bf ldsB(const unsigned short* xs, int rowBase,
                                      int q, int tl, int hi) {
  int base = rowBase + tl * CH + q * 32 + hi * 8;
  Frag u;
  u.q[0] = *(const v4u*)(xs + base);
  u.q[1] = *(const v4u*)(xs + base + 16);
  return u.v;
}

// async global->LDS 16B copy (CDNA5 GLOBAL_LOAD_ASYNC_TO_LDS_B128, GVS mode)
__device__ __forceinline__ void async_g2l_b128(unsigned lds_addr,
                                               unsigned goffset,
                                               const void* sbase) {
  asm volatile("global_load_async_to_lds_b128 %0, %1, %2"
               :
               : "v"(lds_addr), "v"(goffset), "s"(sbase)
               : "memory");
}
__device__ __forceinline__ void wait_async0() {
  asm volatile("s_wait_asynccnt 0x0" ::: "memory");
}

// ---------------------------------------------------------------------------
// Weight repack kernels: fp32 [O][I][K] -> per-lane-contiguous WMMA A fragments
// layout: [...][m_tile][lane(32)][e(16)] bf16 ; cout = 16*m + (lane&15)
// cin = 32*q + (e&7) + ((e&8)?16:0) + (lane>>4)*8
// ---------------------------------------------------------------------------
__global__ void k_repack_wn(const float* __restrict__ w, unsigned short* __restrict__ dst) {
  size_t idx = (size_t)blockIdx.x * 256 + threadIdx.x;
  if (idx >= (size_t)30 * 98304) return;
  int e    = (int)(idx % 16);
  int lane = (int)((idx / 16) % 32);
  int m    = (int)((idx / 512) % 16);
  int q    = (int)((idx / 8192) % 4);
  int tap  = (int)((idx / 32768) % 3);
  int li   = (int)(idx / 98304);
  int hi   = lane >> 4;
  int cout = 16 * m + (lane & 15);
  int kl   = (e & 7) + ((e & 8) << 1) + hi * 8;
  int cin  = 32 * q + kl;
  float s  = w[(((size_t)li * 256 + cout) * 128 + cin) * 3 + tap];
  dst[idx] = f2bf(s);
}

__global__ void k_repack_sk(const float* __restrict__ w, unsigned short* __restrict__ dst) {
  size_t idx = (size_t)blockIdx.x * 256 + threadIdx.x;
  if (idx >= (size_t)30 * 16384) return;
  int e    = (int)(idx % 16);
  int lane = (int)((idx / 16) % 32);
  int m    = (int)((idx / 512) % 8);
  int q    = (int)((idx / 4096) % 4);
  int li   = (int)(idx / 16384);
  int hi   = lane >> 4;
  int cout = 16 * m + (lane & 15);
  int cin  = 32 * q + (e & 7) + ((e & 8) << 1) + hi * 8;
  dst[idx] = f2bf(w[((size_t)li * 128 + cout) * 128 + cin]);
}

__global__ void k_repack_o(const float* __restrict__ w1, const float* __restrict__ w2,
                           unsigned short* __restrict__ d1, unsigned short* __restrict__ d2) {
  int idx = blockIdx.x * 256 + threadIdx.x;
  if (idx >= 2 * 16384) return;
  int sel = idx >> 14;
  int j   = idx & 16383;
  int e    = j % 16;
  int lane = (j / 16) % 32;
  int m    = (j / 512) % 8;
  int q    = j / 4096;
  int hi   = lane >> 4;
  int cout = 16 * m + (lane & 15);
  int cin  = 32 * q + (e & 7) + ((e & 8) << 1) + hi * 8;
  const float* w = sel ? w2 : w1;
  unsigned short* d = sel ? d2 : d1;
  d[j] = f2bf(w[cout * 128 + cin]);
}

// ---------------------------------------------------------------------------
// UpsampleNet (tiny: T=64..1024, plain fp32 VALU)
// ---------------------------------------------------------------------------
__global__ void k_in(const float* __restrict__ mel, const float* __restrict__ w,
                     const float* __restrict__ b, float* __restrict__ x) {
  int idx = blockIdx.x * 256 + threadIdx.x;              // [n][t][c], 4*64*128
  if (idx >= NBATCH * 64 * CH) return;
  int c = idx & 127, t = (idx >> 7) & 63, n = idx >> 13;
  float s = b[c];
  const float* mp = mel + (size_t)n * 80 * 64 + t;       // mel [n][ci][t]
  const float* wp = w + (size_t)c * 80;                  // in_w [c][ci][1]
  for (int i = 0; i < 80; ++i) s += mp[(size_t)i * 64] * wp[i];
  x[idx] = s;
}

__global__ void k_pre(const float* __restrict__ xin, const float* __restrict__ w,
                      const float* __restrict__ b, float* __restrict__ xout, int dil) {
  int idx = blockIdx.x * 256 + threadIdx.x;              // 4*64*128
  if (idx >= NBATCH * 64 * CH) return;
  int c = idx & 127, t = (idx >> 7) & 63, n = idx >> 13;
  float s = b[c];
  const float* wr = w + (size_t)c * 128 * 3;             // [O][I][3]
  #pragma unroll
  for (int tap = 0; tap < 3; ++tap) {
    int ti = t + (tap - 1) * dil;
    if (ti < 0 || ti >= 64) continue;
    const float* xr = xin + (((size_t)n * 64 + ti) << 7);
    for (int i = 0; i < 128; ++i) s += xr[i] * wr[i * 3 + tap];
  }
  s = s > 0.f ? s : 0.1f * s;                            // leaky relu
  xout[idx] = (s + xin[idx]) * 0.70710678118654752f;
}

// conv_transpose k=4, stride=4, VALID: y[n][4q+r][o] = sum_i w[o][i][3-r]*x[n][q][i]+b
__global__ void k_up(const float* __restrict__ xin, const float* __restrict__ w,
                     const float* __restrict__ b, float* __restrict__ xout, int Lin) {
  int Lout = Lin * 4;
  int idx = blockIdx.x * 256 + threadIdx.x;
  if (idx >= NBATCH * Lout * CH) return;
  int c = idx & 127;
  int t = (idx >> 7) % Lout;
  int n = (idx >> 7) / Lout;
  int q = t >> 2, tap = 3 - (t & 3);
  float s = b[c];
  const float* xr = xin + (((size_t)n * Lin + q) << 7);
  const float* wr = w + (size_t)c * 128 * 4 + tap;       // [O][I][4]
  for (int i = 0; i < 128; ++i) s += xr[i] * wr[i * 4];
  xout[idx] = s > 0.f ? s : 0.1f * s;
}

__global__ void k_repeat(const float* __restrict__ x1024, unsigned short* __restrict__ xbf,
                         float* __restrict__ skips) {
  size_t idx = (size_t)blockIdx.x * 256 + threadIdx.x;   // 4*16384*128
  if (idx >= (size_t)NBATCH * T_FIN * CH) return;
  int c = (int)(idx & 127);
  size_t t = (idx >> 7) & 16383;
  size_t n = idx >> 21;
  xbf[idx]   = f2bf(x1024[((n * 1024 + (t >> 4)) << 7) + c]);
  skips[idx] = 0.f;
}

// ---------------------------------------------------------------------------
// WaveNet dilated conv 128->256 + fused gate  (dominant kernel, WMMA bf16)
// block = 256 thr = 8 waves, one (n, 64-time tile) per block (4 N-tiles).
// wave w: cout tile w (tanh half) + tile w+8 (sigmoid half); each A fragment
// reused across 4 N-tiles -> 96 WMMAs per wave, 4x less weight traffic.
// ---------------------------------------------------------------------------
__global__ __launch_bounds__(256) void k_wn_gate(
    const unsigned short* __restrict__ xbf,
    const unsigned short* __restrict__ wpack,   // this layer: [tap][q][m16][lane][16]
    const float* __restrict__ wbias,            // this layer: 256 floats
    unsigned short* __restrict__ gated,
    int dil) {
  __shared__ unsigned short xs[3 * TBLK * CH];  // 48 KB: 3 strips x 64t x 128c
  const int n  = blockIdx.x >> 8;               // 256 time-blocks per batch
  const int t0 = (blockIdx.x & 255) << 6;
  const int tid = threadIdx.x;

  // warm L2/WGP$ with this layer's packed weights
  __builtin_prefetch((const char*)wpack + tid * 256, 0, 1);

  const unsigned short* xbase = xbf + ((size_t)n * T_FIN << 7);
  const unsigned lds_base = (unsigned)(size_t)(void*)xs;

  // stage 3 strips via async global->LDS; zero-fill padded rows
  // 3 strips * 64 rows * 16 chunks(16B) = 3072 chunks, 12 per thread
  #pragma unroll
  for (int it = 0; it < 12; ++it) {
    int chunk = tid + it * 256;
    int j   = chunk >> 10;                      // strip 0..2
    int rem = chunk & 1023;
    int row = rem >> 4;                         // 0..63
    int c16 = rem & 15;
    int t_in = t0 + row + (j - 1) * dil;
    int eoff = (j * TBLK + row) * CH + c16 * 8; // element offset in xs
    if (t_in >= 0 && t_in < T_FIN) {
      async_g2l_b128(lds_base + (unsigned)(eoff * 2),
                     (unsigned)(((t_in << 7) + c16 * 8) * 2), xbase);
    } else {
      v4u z = {0u, 0u, 0u, 0u};
      *(v4u*)(xs + eoff) = z;
    }
  }
  wait_async0();
  __syncthreads();

  const int w = tid >> 5, lane = tid & 31;
  const int tl = lane & 15, hi = lane >> 4;

  v8f at[4], as_[4];
  #pragma unroll
  for (int nt = 0; nt < 4; ++nt) {
    at[nt]  = (v8f){0, 0, 0, 0, 0, 0, 0, 0};
    as_[nt] = (v8f){0, 0, 0, 0, 0, 0, 0, 0};
  }

  #pragma unroll
  for (int tap = 0; tap < 3; ++tap) {
    #pragma unroll
    for (int q = 0; q < 4; ++q) {
      size_t ab  = ((((size_t)tap * 4 + q) * 16 + w) * 32 + lane) * 16;
      size_t ab2 = ((((size_t)tap * 4 + q) * 16 + (w + 8)) * 32 + lane) * 16;
      v16bf a_t = *(const v16bf*)(wpack + ab);
      v16bf a_s = *(const v16bf*)(wpack + ab2);
      #pragma unroll
      for (int nt = 0; nt < 4; ++nt) {
        v16bf bm = ldsB(xs, (tap * TBLK + nt * 16) * CH, q, tl, hi);
        at[nt]  = wmma_bf16(a_t, bm, at[nt]);
        as_[nt] = wmma_bf16(a_s, bm, as_[nt]);
      }
    }
  }

  const int cb = (w << 4) + (hi << 3);          // channel base (M = r + 8*hi)
  float bt[8], bs[8];
  #pragma unroll
  for (int r = 0; r < 8; ++r) {
    bt[r] = wbias[cb + r];
    bs[r] = wbias[128 + cb + r];
  }
  #pragma unroll
  for (int nt = 0; nt < 4; ++nt) {
    float g8[8];
    #pragma unroll
    for (int r = 0; r < 8; ++r) {
      float gt = at[nt][r] + bt[r];
      float gs = as_[nt][r] + bs[r];
      g8[r] = tanhf(gt) * (1.f / (1.f + expf(-gs)));
    }
    int t = t0 + nt * 16 + tl;
    *(v4u*)(gated + ((size_t)(n * T_FIN + t) << 7) + cb) = pack8(g8);
  }
}

// ---------------------------------------------------------------------------
// Skip 1x1 conv 128->128 + skips += s  + x = bf16(s + x_old)   (WMMA bf16)
// ---------------------------------------------------------------------------
__global__ __launch_bounds__(256) void k_wn_skip(
    const unsigned short* __restrict__ gated,
    const unsigned short* __restrict__ skpack,  // this layer: [q][m8][lane][16]
    const float* __restrict__ skbias,           // this layer: 128 floats
    float* __restrict__ skips,
    unsigned short* __restrict__ xbf) {
  __shared__ unsigned short gs[16 * CH];        // 4 KB
  const int n  = blockIdx.x >> 10;
  const int t0 = (blockIdx.x & 1023) << 4;
  const int tid = threadIdx.x;
  {
    int row = tid >> 4, c16 = tid & 15;
    const unsigned short* gbase = gated + ((size_t)n * T_FIN << 7);
    async_g2l_b128((unsigned)(size_t)(void*)gs + (unsigned)((row * CH + c16 * 8) * 2),
                   (unsigned)(((((unsigned)(t0 + row)) << 7) + c16 * 8) * 2), gbase);
  }
  wait_async0();
  __syncthreads();

  const int w = tid >> 5, lane = tid & 31;
  const int tl = lane & 15, hi = lane >> 4;

  v8f acc = {0, 0, 0, 0, 0, 0, 0, 0};
  #pragma unroll
  for (int q = 0; q < 4; ++q) {
    v16bf bm = ldsB(gs, 0, q, tl, hi);
    size_t ab = (((size_t)q * 8 + w) * 32 + lane) * 16;
    v16bf a = *(const v16bf*)(skpack + ab);
    acc = wmma_bf16(a, bm, acc);
  }

  const int t  = t0 + tl;
  const int cb = (w << 4) + (hi << 3);
  size_t off = ((size_t)(n * T_FIN + t) << 7) + cb;

  float s8[8];
  #pragma unroll
  for (int r = 0; r < 8; ++r) s8[r] = acc[r] + skbias[cb + r];

  float* sp = skips + off;
  v4f s0 = *(v4f*)sp, s1 = *(v4f*)(sp + 4);
  s0.x += s8[0]; s0.y += s8[1]; s0.z += s8[2]; s0.w += s8[3];
  s1.x += s8[4]; s1.y += s8[5]; s1.z += s8[6]; s1.w += s8[7];
  *(v4f*)sp = s0; *(v4f*)(sp + 4) = s1;

  float xo[8];
  unpack8(*(const v4u*)(xbf + off), xo);
  float xn[8];
  #pragma unroll
  for (int r = 0; r < 8; ++r) xn[r] = s8[r] + xo[r];
  *(v4u*)(xbf + off) = pack8(xn);
}

// ---------------------------------------------------------------------------
// Output 1x1 conv 128->128 + ReLU.
// src_is_f32=1: src = fp32 skips, relu+cvt staged through VGPRs.
// src_is_f32=0: src = bf16 (already post-ReLU) -> pure async copy to LDS.
// ---------------------------------------------------------------------------
__global__ __launch_bounds__(256) void k_out1x1(
    const void* __restrict__ src, int src_is_f32,
    const unsigned short* __restrict__ wpack,   // [q][m8][lane][16]
    const float* __restrict__ bias,
    unsigned short* __restrict__ dst) {
  __shared__ unsigned short ys[16 * CH];
  const int n  = blockIdx.x >> 10;
  const int t0 = (blockIdx.x & 1023) << 4;
  const int tid = threadIdx.x;
  {
    int row = tid >> 4, c16 = tid & 15;
    size_t off = ((size_t)(n * T_FIN + t0 + row) << 7) + c16 * 8;
    if (src_is_f32) {
      const float* sp = (const float*)src + off;
      float f[8];
      v4f a = *(const v4f*)sp, b = *(const v4f*)(sp + 4);
      f[0] = a.x; f[1] = a.y; f[2] = a.z; f[3] = a.w;
      f[4] = b.x; f[5] = b.y; f[6] = b.z; f[7] = b.w;
      #pragma unroll
      for (int r = 0; r < 8; ++r) f[r] = f[r] > 0.f ? f[r] : 0.f;
      *(v4u*)(ys + row * CH + c16 * 8) = pack8(f);
    } else {
      const unsigned short* gbase = (const unsigned short*)src + ((size_t)n * T_FIN << 7);
      async_g2l_b128((unsigned)(size_t)(void*)ys + (unsigned)((row * CH + c16 * 8) * 2),
                     (unsigned)(((((unsigned)(t0 + row)) << 7) + c16 * 8) * 2), gbase);
    }
  }
  wait_async0();
  __syncthreads();

  const int w = tid >> 5, lane = tid & 31;
  const int tl = lane & 15, hi = lane >> 4;

  v8f acc = {0, 0, 0, 0, 0, 0, 0, 0};
  #pragma unroll
  for (int q = 0; q < 4; ++q) {
    v16bf bm = ldsB(ys, 0, q, tl, hi);
    size_t ab = (((size_t)q * 8 + w) * 32 + lane) * 16;
    v16bf a = *(const v16bf*)(wpack + ab);
    acc = wmma_bf16(a, bm, acc);
  }

  const int t  = t0 + tl;
  const int cb = (w << 4) + (hi << 3);
  float f[8];
  #pragma unroll
  for (int r = 0; r < 8; ++r) {
    float v = acc[r] + bias[cb + r];
    f[r] = v > 0.f ? v : 0.f;                   // relu (out1 and out2 both use it)
  }
  *(v4u*)(dst + ((size_t)(n * T_FIN + t) << 7) + cb) = pack8(f);
}

__global__ void k_out3(const unsigned short* __restrict__ y,
                       const float* __restrict__ w3, const float* __restrict__ b3,
                       float* __restrict__ out) {
  int idx = blockIdx.x * 256 + threadIdx.x;     // n*T + t
  if (idx >= NBATCH * T_FIN) return;
  const unsigned short* yp = y + (size_t)idx * CH;
  float s = b3[0];
  for (int c = 0; c < 128; ++c) s += bf2f(yp[c]) * w3[c];
  out[idx] = tanhf(s);
}

// ---------------------------------------------------------------------------
// Host driver
// ---------------------------------------------------------------------------
extern "C" void kernel_launch(void* const* d_in, const int* in_sizes, int n_in,
                              void* d_out, int out_size, void* d_ws, size_t ws_size,
                              hipStream_t stream) {
  (void)in_sizes; (void)n_in; (void)out_size; (void)ws_size;
  const float* mel    = (const float*)d_in[0];
  const float* in_w   = (const float*)d_in[1];
  const float* in_b   = (const float*)d_in[2];
  const float* pre_ws = (const float*)d_in[3];
  const float* pre_bs = (const float*)d_in[4];
  const float* up1_w  = (const float*)d_in[5];
  const float* up1_b  = (const float*)d_in[6];
  const float* up2_w  = (const float*)d_in[7];
  const float* up2_b  = (const float*)d_in[8];
  const float* wn_ws  = (const float*)d_in[9];
  const float* wn_bs  = (const float*)d_in[10];
  const float* sk_ws  = (const float*)d_in[11];
  const float* sk_bs  = (const float*)d_in[12];
  const float* out1_w = (const float*)d_in[13];
  const float* out1_b = (const float*)d_in[14];
  const float* out2_w = (const float*)d_in[15];
  const float* out2_b = (const float*)d_in[16];
  const float* out3_w = (const float*)d_in[17];
  const float* out3_b = (const float*)d_in[18];

  char* ws = (char*)d_ws;
  float*          x64    = (float*)(ws + OFF_X64);
  float*          tmp64  = (float*)(ws + OFF_TMP64);
  float*          x256   = (float*)(ws + OFF_X256);
  float*          x1024  = (float*)(ws + OFF_X1024);
  unsigned short* xbf    = (unsigned short*)(ws + OFF_XBF);
  unsigned short* gated  = (unsigned short*)(ws + OFF_GATED);
  float*          skips  = (float*)(ws + OFF_SKIPS);
  unsigned short* wnp    = (unsigned short*)(ws + OFF_WNP);
  unsigned short* skp    = (unsigned short*)(ws + OFF_SKP);
  unsigned short* o1p    = (unsigned short*)(ws + OFF_O1P);
  unsigned short* o2p    = (unsigned short*)(ws + OFF_O2P);

  // weight repack (deterministic, every call)
  k_repack_wn<<<11520, 256, 0, stream>>>(wn_ws, wnp);
  k_repack_sk<<<1920, 256, 0, stream>>>(sk_ws, skp);
  k_repack_o<<<128, 256, 0, stream>>>(out1_w, out2_w, o1p, o2p);

  // UpsampleNet
  k_in<<<128, 256, 0, stream>>>(mel, in_w, in_b, x64);
  float* a = x64; float* b = tmp64;
  for (int i = 0; i < 5; ++i) {
    k_pre<<<128, 256, 0, stream>>>(a, pre_ws + (size_t)i * 128 * 128 * 3,
                                   pre_bs + i * 128, b, 1 << i);
    float* t = a; a = b; b = t;
  }
  k_up<<<512, 256, 0, stream>>>(a, up1_w, up1_b, x256, 64);
  k_up<<<2048, 256, 0, stream>>>(x256, up2_w, up2_b, x1024, 256);
  k_repeat<<<32768, 256, 0, stream>>>(x1024, xbf, skips);

  // WaveNet: 30 layers
  for (int li = 0; li < 30; ++li) {
    int dil = 1 << (li % 10);
    k_wn_gate<<<1024, 256, 0, stream>>>(xbf, wnp + (size_t)li * 98304,
                                        wn_bs + li * 256, gated, dil);
    k_wn_skip<<<4096, 256, 0, stream>>>(gated, skp + (size_t)li * 16384,
                                        sk_bs + li * 128, skips, xbf);
  }

  // Output stack: relu(skips) -> conv+relu -> conv+relu -> conv -> tanh
  k_out1x1<<<4096, 256, 0, stream>>>(skips, 1, o1p, out1_b, gated);
  k_out1x1<<<4096, 256, 0, stream>>>(gated, 0, o2p, out2_b, xbf);
  k_out3<<<256, 256, 0, stream>>>(xbf, out3_w, out3_b, (float*)d_out);
}
